// SelfAttentionModule_65481071401272
// MI455X (gfx1250) — matrix operand order repaired and checked by reading, not compiled, not run on hardware
//
#include <hip/hip_runtime.h>

// ---------------------------------------------------------------------------
// CDNA5 (gfx1250) fused self-attention block, bf16 WMMA (16x16x32, f32 acc)
// Two-stage ping-pong pipelined GEMMs + flash attention (no buffer copies).
// ---------------------------------------------------------------------------

typedef __bf16 bf16_t;
typedef __attribute__((ext_vector_type(16))) __bf16 v16bf;
typedef __attribute__((ext_vector_type(8)))  __bf16 v8bf;
typedef __attribute__((ext_vector_type(8)))  float  v8f;

#define D_IN   1024
#define D_QKV  3072
#define S_LEN  2048
#define H_NUM  16
#define HD     64

__device__ __forceinline__ bf16_t f2bf(float f) {
  unsigned int u = __builtin_bit_cast(unsigned int, f);
  unsigned int r = (u + 0x7FFFu + ((u >> 16) & 1u)) >> 16;
  return __builtin_bit_cast(__bf16, static_cast<unsigned short>(r));
}

__device__ __forceinline__ v8f wmma_bf16(v16bf a, v16bf b, v8f c) {
  // D = A(16x32 bf16) * B(32x16 bf16) + C(16x16 f32)
  return __builtin_amdgcn_wmma_f32_16x16x32_bf16(
      /*neg_a=*/false, a, /*neg_b=*/false, b,
      /*c_mod=*/(short)0, c, /*reuse_a=*/false, /*reuse_b=*/false);
}

__device__ __forceinline__ v16bf pack16(v8bf lo, v8bf hi) {
  v16bf a;
#pragma unroll
  for (int i = 0; i < 8; ++i) { a[i] = lo[i]; a[i + 8] = hi[i]; }
  return a;
}

// ---------------------------------------------------------------------------
// Kernel 0: transpose fp32 weight [1024 x Ncols] -> bf16 [Ncols x 1024]
// ---------------------------------------------------------------------------
__global__ void __launch_bounds__(256) wtrans_kernel(
    const float* __restrict__ W, bf16_t* __restrict__ Wt, int Ncols) {
  int idx = blockIdx.x * 256 + threadIdx.x;
  if (idx >= 1024 * Ncols) return;
  int n = idx >> 10;
  int k = idx & 1023;
  Wt[idx] = f2bf(W[(size_t)k * Ncols + n]);
}

// ---------------------------------------------------------------------------
// Kernel 0b: x fp32 -> bf16
// ---------------------------------------------------------------------------
__global__ void __launch_bounds__(256) xconv_kernel(
    const float* __restrict__ x, bf16_t* __restrict__ xb) {
  size_t idx = (size_t)blockIdx.x * 256 + threadIdx.x;
  xb[idx] = f2bf(x[idx]);
}

// ---------------------------------------------------------------------------
// Kernel 1: QKV = x @ W_qkv + b_qkv.  Wave = 16x64 tile, 2-stage ping-pong.
// Q,K -> [B,H,S,Hd] bf16 ; V -> transposed [B,H,Hd,S] bf16.
// ---------------------------------------------------------------------------
__global__ void __launch_bounds__(128) qkv_proj_kernel(
    const bf16_t* __restrict__ Xb, const bf16_t* __restrict__ Wt,
    const float* __restrict__ bias,
    bf16_t* __restrict__ Qo, bf16_t* __restrict__ Ko, bf16_t* __restrict__ Vt) {
  const int lane = threadIdx.x & 31;
  const int wave = __builtin_amdgcn_readfirstlane((int)(threadIdx.x >> 5));
  const int l16  = lane & 15;
  const int hi   = lane >> 4;
  const int m0   = blockIdx.y * 16;
  const int n0   = blockIdx.x * 256 + wave * 64;   // scalar

  const bf16_t* arow = Xb + (size_t)(m0 + l16) * D_IN + hi * 16;
  const bf16_t* brow[4];
#pragma unroll
  for (int t = 0; t < 4; ++t)
    brow[t] = Wt + (size_t)(n0 + t * 16 + l16) * D_IN + hi * 16;

  v8f acc[4] = {};
  v16bf aA, bA[4], aB, bB[4];

  auto loadStage = [&](int k, v16bf& a, v16bf* b) {
    a = pack16(*(const v8bf*)(arow + k), *(const v8bf*)(arow + k + 8));
#pragma unroll
    for (int t = 0; t < 4; ++t) b[t] = *(const v16bf*)(brow[t] + k);
  };

  loadStage(0, aA, bA);
  for (int k0 = 0; k0 < D_IN; k0 += 64) {
    loadStage(k0 + 32, aB, bB);                     // k0+32 always < D_IN
    __builtin_prefetch(brow[0] + ((k0 + 128) & (D_IN - 1)), 0, 2);
#pragma unroll
    for (int t = 0; t < 4; ++t) acc[t] = wmma_bf16(aA, bA[t], acc[t]);
    loadStage((k0 + 64) & (D_IN - 1), aA, bA);      // wraps on last iter
#pragma unroll
    for (int t = 0; t < 4; ++t) acc[t] = wmma_bf16(aB, bB[t], acc[t]);
  }

  // epilogue: n0 is scalar -> uniform routing branch, hd = t*16+l16
  const int h = (n0 & 1023) >> 6;
  if (n0 < 1024) {
#pragma unroll
    for (int t = 0; t < 4; ++t) {
      const float bv = bias[n0 + t * 16 + l16];
#pragma unroll
      for (int r = 0; r < 8; ++r) {
        const int m = m0 + r + hi * 8, bi = m >> 11, s = m & (S_LEN - 1);
        Qo[(((size_t)(bi * H_NUM + h)) * S_LEN + s) * HD + t * 16 + l16] =
            f2bf(acc[t][r] + bv);
      }
    }
  } else if (n0 < 2048) {
#pragma unroll
    for (int t = 0; t < 4; ++t) {
      const float bv = bias[n0 + t * 16 + l16];
#pragma unroll
      for (int r = 0; r < 8; ++r) {
        const int m = m0 + r + hi * 8, bi = m >> 11, s = m & (S_LEN - 1);
        Ko[(((size_t)(bi * H_NUM + h)) * S_LEN + s) * HD + t * 16 + l16] =
            f2bf(acc[t][r] + bv);
      }
    }
  } else {
#pragma unroll
    for (int t = 0; t < 4; ++t) {
      const float bv = bias[n0 + t * 16 + l16];
#pragma unroll
      for (int r = 0; r < 8; ++r) {
        const int m = m0 + r + hi * 8, bi = m >> 11, s = m & (S_LEN - 1);
        Vt[(((size_t)(bi * H_NUM + h)) * HD + t * 16 + l16) * S_LEN + s] =
            f2bf(acc[t][r] + bv);
      }
    }
  }
}

// ---------------------------------------------------------------------------
// Kernel 2: flash attention.  One wave per 16-query tile; two 32-key stages
// per loop iteration, K-fragments ping-pong (no rotation copies).
// ---------------------------------------------------------------------------
__global__ void __launch_bounds__(128) attn_kernel(
    const bf16_t* __restrict__ Q, const bf16_t* __restrict__ K,
    const bf16_t* __restrict__ Vt, bf16_t* __restrict__ ctx) {
  __shared__ __attribute__((aligned(16))) float  sbuf[4][16][32];
  __shared__ __attribute__((aligned(16))) bf16_t pbuf[4][16][32];
  __shared__ __attribute__((aligned(16))) float  abuf[4][16];

  const int lane = threadIdx.x & 31;
  const int wv   = __builtin_amdgcn_readfirstlane((int)(threadIdx.x >> 5));
  const int l16  = lane & 15;
  const int hi   = lane >> 4;

  const int tile = __builtin_amdgcn_readfirstlane(blockIdx.x * 4) + wv;
  const int bh   = tile >> 7;                  // scalar (b*H + h)
  const int s0   = (tile & 127) * 16;          // scalar

  const bf16_t* Kbase = K  + (size_t)bh * S_LEN * HD;
  const bf16_t* Vbase = Vt + (size_t)bh * HD * S_LEN;

  // Q A-fragments for hd 0..31 and 32..63
  const bf16_t* qrow = Q + ((size_t)bh * S_LEN + s0 + l16) * HD + hi * 16;
  v16bf a0 = pack16(*(const v8bf*)(qrow),      *(const v8bf*)(qrow + 8));
  v16bf a1 = pack16(*(const v8bf*)(qrow + 32), *(const v8bf*)(qrow + 40));

  v8f o[4] = {};
  float m_row = -3.0e38f, l_row = 0.0f;

  // Load 4 K B-fragments covering keys [key0, key0+32)
  auto loadK32 = [&](int key0, v16bf* kf) {
    const bf16_t* kp0 = Kbase + (size_t)(key0 + l16) * HD + hi * 16;
    const bf16_t* kp1 = kp0 + 16 * HD;
    kf[0] = *(const v16bf*)(kp0);
    kf[1] = *(const v16bf*)(kp0 + 32);
    kf[2] = *(const v16bf*)(kp1);
    kf[3] = *(const v16bf*)(kp1 + 32);
  };

  // One 32-key flash-attention step
  auto step = [&](const v16bf* kf, int j) {
    v16bf vb[4];
#pragma unroll
    for (int t = 0; t < 4; ++t)
      vb[t] = *(const v16bf*)(Vbase + (size_t)(t * 16 + l16) * S_LEN + j + hi * 16);

    v8f sc0 = {}, sc1 = {};
    sc0 = wmma_bf16(a0, kf[0], sc0);
    sc0 = wmma_bf16(a1, kf[1], sc0);
    sc1 = wmma_bf16(a0, kf[2], sc1);
    sc1 = wmma_bf16(a1, kf[3], sc1);

#pragma unroll
    for (int r = 0; r < 8; ++r) {
      const int row = r + hi * 8;
      sbuf[wv][row][l16]      = sc0[r] * 0.125f;   // 1/sqrt(64)
      sbuf[wv][row][16 + l16] = sc1[r] * 0.125f;
    }
    asm volatile("s_wait_dscnt 0x0" ::: "memory");  // wave-local LDS sync

    if (lane < 16) {                                // online softmax, 1 row/lane
      const int row = lane;
      const float4* srow = (const float4*)&sbuf[wv][row][0];
      float4 sv[8];
#pragma unroll
      for (int c = 0; c < 8; ++c) sv[c] = srow[c];
      float mj = -3.0e38f;
#pragma unroll
      for (int c = 0; c < 8; ++c)
        mj = fmaxf(mj, fmaxf(fmaxf(sv[c].x, sv[c].y), fmaxf(sv[c].z, sv[c].w)));
      float newm = fmaxf(m_row, mj);
      float al   = __expf(m_row - newm);
      float sum  = 0.0f;
#pragma unroll
      for (int c = 0; c < 8; ++c) {
        float p0 = __expf(sv[c].x - newm), p1 = __expf(sv[c].y - newm);
        float p2 = __expf(sv[c].z - newm), p3 = __expf(sv[c].w - newm);
        pbuf[wv][row][c * 4 + 0] = f2bf(p0);
        pbuf[wv][row][c * 4 + 1] = f2bf(p1);
        pbuf[wv][row][c * 4 + 2] = f2bf(p2);
        pbuf[wv][row][c * 4 + 3] = f2bf(p3);
        sum += (p0 + p1) + (p2 + p3);
      }
      l_row = l_row * al + sum;
      m_row = newm;
      abuf[wv][row] = al;
    }
    asm volatile("s_wait_dscnt 0x0" ::: "memory");

    float4 al0 = *(const float4*)(&abuf[wv][hi * 8]);
    float4 al1 = *(const float4*)(&abuf[wv][hi * 8 + 4]);
    float alr[8] = {al0.x, al0.y, al0.z, al0.w, al1.x, al1.y, al1.z, al1.w};
#pragma unroll
    for (int t = 0; t < 4; ++t)
#pragma unroll
      for (int r = 0; r < 8; ++r) o[t][r] *= alr[r];

    const bf16_t* pr = &pbuf[wv][l16][0];
    v16bf pa = pack16(*(const v8bf*)(pr + hi * 16), *(const v8bf*)(pr + hi * 16 + 8));

#pragma unroll
    for (int t = 0; t < 4; ++t) o[t] = wmma_bf16(pa, vb[t], o[t]);
  };

  v16bf kfA[4], kfB[4];
  loadK32(0, kfA);
  for (int j = 0; j < S_LEN; j += 64) {
    loadK32(j + 32, kfB);                        // always in range
    step(kfA, j);
    loadK32((j + 64) & (S_LEN - 1), kfA);        // wraps on last iter
    step(kfB, j + 32);
  }

  // ---- final 1/l normalization and ctx store [B,S,D] bf16 ----------------
  if (lane < 16) abuf[wv][lane] = (l_row > 0.0f) ? (1.0f / l_row) : 0.0f;
  asm volatile("s_wait_dscnt 0x0" ::: "memory");
  float4 li0 = *(const float4*)(&abuf[wv][hi * 8]);
  float4 li1 = *(const float4*)(&abuf[wv][hi * 8 + 4]);
  float lir[8] = {li0.x, li0.y, li0.z, li0.w, li1.x, li1.y, li1.z, li1.w};

  const int b = bh >> 4, h = bh & 15;
#pragma unroll
  for (int t = 0; t < 4; ++t) {
    const int hd = t * 16 + l16;
#pragma unroll
    for (int r = 0; r < 8; ++r) {
      const int s = s0 + r + hi * 8;
      ctx[((size_t)(b * S_LEN + s)) * D_IN + h * HD + hd] = f2bf(o[t][r] * lir[r]);
    }
  }
}

// ---------------------------------------------------------------------------
// Kernel 3: Y = ctx @ W_out + b_out + x   (pre-LayerNorm, fp32), ping-pong
// ---------------------------------------------------------------------------
__global__ void __launch_bounds__(128) outproj_kernel(
    const bf16_t* __restrict__ ctx, const bf16_t* __restrict__ Wt,
    const float* __restrict__ bias, const float* __restrict__ x,
    float* __restrict__ y) {
  const int lane = threadIdx.x & 31;
  const int wave = __builtin_amdgcn_readfirstlane((int)(threadIdx.x >> 5));
  const int l16  = lane & 15;
  const int hi   = lane >> 4;
  const int m0   = blockIdx.y * 16;
  const int n0   = blockIdx.x * 256 + wave * 64;

  const bf16_t* arow = ctx + (size_t)(m0 + l16) * D_IN + hi * 16;
  const bf16_t* brow[4];
#pragma unroll
  for (int t = 0; t < 4; ++t)
    brow[t] = Wt + (size_t)(n0 + t * 16 + l16) * D_IN + hi * 16;

  v8f acc[4] = {};
  v16bf aA, bA[4], aB, bB[4];

  auto loadStage = [&](int k, v16bf& a, v16bf* b) {
    a = pack16(*(const v8bf*)(arow + k), *(const v8bf*)(arow + k + 8));
#pragma unroll
    for (int t = 0; t < 4; ++t) b[t] = *(const v16bf*)(brow[t] + k);
  };

  loadStage(0, aA, bA);
  for (int k0 = 0; k0 < D_IN; k0 += 64) {
    loadStage(k0 + 32, aB, bB);
    __builtin_prefetch(brow[0] + ((k0 + 128) & (D_IN - 1)), 0, 2);
#pragma unroll
    for (int t = 0; t < 4; ++t) acc[t] = wmma_bf16(aA, bA[t], acc[t]);
    loadStage((k0 + 64) & (D_IN - 1), aA, bA);
#pragma unroll
    for (int t = 0; t < 4; ++t) acc[t] = wmma_bf16(aB, bB[t], acc[t]);
  }

#pragma unroll
  for (int t = 0; t < 4; ++t) {
    const int ncol = n0 + t * 16 + l16;
    const float bv = bias[ncol];
#pragma unroll
    for (int r = 0; r < 8; ++r) {
      const int m = m0 + r + hi * 8;
      y[(size_t)m * D_IN + ncol] = acc[t][r] + bv + x[(size_t)m * D_IN + ncol];
    }
  }
}

// ---------------------------------------------------------------------------
// Kernel 4: row LayerNorm -> fp32 output
// ---------------------------------------------------------------------------
__global__ void __launch_bounds__(256) ln_kernel(
    const float* __restrict__ y, const float* __restrict__ gamma,
    const float* __restrict__ beta, float* __restrict__ out) {
  __shared__ float red[256];
  const int tid = threadIdx.x;
  const size_t row = blockIdx.x;
  const float* yr = y + row * D_IN;

  float s = 0.0f;
  for (int i = tid; i < D_IN; i += 256) s += yr[i];
  red[tid] = s; __syncthreads();
  for (int st = 128; st > 0; st >>= 1) {
    if (tid < st) red[tid] += red[tid + st];
    __syncthreads();
  }
  const float mu = red[0] * (1.0f / D_IN);
  __syncthreads();

  float v = 0.0f;
  for (int i = tid; i < D_IN; i += 256) { float d = yr[i] - mu; v += d * d; }
  red[tid] = v; __syncthreads();
  for (int st = 128; st > 0; st >>= 1) {
    if (tid < st) red[tid] += red[tid + st];
    __syncthreads();
  }
  const float rs = rsqrtf(red[0] * (1.0f / D_IN) + 1e-5f);

  for (int i = tid; i < D_IN; i += 256)
    out[row * D_IN + i] = (yr[i] - mu) * rs * gamma[i] + beta[i];
}

// ---------------------------------------------------------------------------
// Launch
// ---------------------------------------------------------------------------
extern "C" void kernel_launch(void* const* d_in, const int* in_sizes, int n_in,
                              void* d_out, int out_size, void* d_ws, size_t ws_size,
                              hipStream_t stream) {
  (void)in_sizes; (void)n_in; (void)out_size; (void)ws_size;
  const float* x     = (const float*)d_in[0];
  const float* W_qkv = (const float*)d_in[1];
  const float* b_qkv = (const float*)d_in[2];
  const float* W_out = (const float*)d_in[3];
  const float* b_out = (const float*)d_in[4];
  const float* gamma = (const float*)d_in[5];
  const float* beta  = (const float*)d_in[6];
  float* out = (float*)d_out;

  char* ws = (char*)d_ws;
  bf16_t* Wt_qkv = (bf16_t*)(ws + 0);           //  6,291,456 B
  bf16_t* Wt_out = (bf16_t*)(ws + 6291456);     //  2,097,152 B
  bf16_t* Xb     = (bf16_t*)(ws + 8388608);     // 16,777,216 B  [B*S, D]
  bf16_t* Qb     = (bf16_t*)(ws + 25165824);    // 16,777,216 B  [B,H,S,Hd]
  bf16_t* Kb     = (bf16_t*)(ws + 41943040);    // 16,777,216 B  [B,H,S,Hd]
  bf16_t* Vtb    = (bf16_t*)(ws + 58720256);    // 16,777,216 B  [B,H,Hd,S]
  bf16_t* Ctx    = (bf16_t*)(ws + 75497472);    // 16,777,216 B  [B,S,D]
  float*  Y      = (float*)(ws + 92274688);     // 33,554,432 B  (~120 MB total)

  wtrans_kernel<<<(1024 * 3072) / 256, 256, 0, stream>>>(W_qkv, Wt_qkv, 3072);
  wtrans_kernel<<<(1024 * 1024) / 256, 256, 0, stream>>>(W_out, Wt_out, 1024);
  xconv_kernel<<<(8192 * 1024) / 256, 256, 0, stream>>>(x, Xb);
  qkv_proj_kernel<<<dim3(12, 512), 128, 0, stream>>>(Xb, Wt_qkv, b_qkv, Qb, Kb, Vtb);
  attn_kernel<<<2048, 128, 0, stream>>>(Qb, Kb, Vtb, Ctx);
  outproj_kernel<<<dim3(4, 512), 128, 0, stream>>>(Ctx, Wt_out, b_out, x, Y);
  ln_kernel<<<8192, 256, 0, stream>>>(Y, gamma, beta, out);
}
